// TransformerDecoderLayer_48447231098916
// MI455X (gfx1250) — matrix operand order repaired and checked by reading, not compiled, not run on hardware
//
#include <hip/hip_runtime.h>

// ---------------------------------------------------------------------------
// Types for CDNA5 WMMA (wave32): v_wmma_f32_16x16x32_bf16
// ---------------------------------------------------------------------------
typedef __attribute__((ext_vector_type(16))) __bf16 bf16x16;
typedef __attribute__((ext_vector_type(8)))  float  f32x8;
typedef __attribute__((ext_vector_type(4)))  unsigned int u32x4;

union Frag {
    bf16x16 v;
    u32x4   q[2];
};

__device__ __forceinline__ unsigned short f2bf(float f) {
    union { float f; unsigned u; } x;
    x.f = f;
    unsigned r = x.u + 0x7FFFu + ((x.u >> 16) & 1u);  // round-to-nearest-even
    return (unsigned short)(r >> 16);
}

// A-matrix 16x32 bf16 fragment: lane m=L&15, h=L>>4 holds K in
// {8h..8h+7} U {16+8h..16+8h+7} (ISA 7.12.2). p -> start of 32-elem K-run.
__device__ __forceinline__ Frag ldA(const unsigned short* p, int hh) {
    Frag f;
    f.q[0] = *(const u32x4*)(p + 8 * hh);
    f.q[1] = *(const u32x4*)(p + 16 + 8 * hh);
    return f;
}
// B-matrix 32x16 bf16 fragment: lane n=L&15, half h holds K in
// {8h..8h+7} U {16+8h..16+8h+7} (same order as A).
// p -> start of that column's 32-elem K-run (K-major / transposed storage).
__device__ __forceinline__ Frag ldB(const unsigned short* p, int hh) {
    Frag f;
    f.q[0] = *(const u32x4*)(p + 8 * hh);
    f.q[1] = *(const u32x4*)(p + 16 + 8 * hh);
    return f;
}

#define WMMA_BF16(a, b, c) \
    __builtin_amdgcn_wmma_f32_16x16x32_bf16(false, (a), false, (b), (short)0, (c), false, false)

// CDNA5 async global->LDS copy (ASYNCcnt-tracked DMA path, GV addressing).
__device__ __forceinline__ void async_load_b128(unsigned lds_off, const void* gaddr) {
    asm volatile("global_load_async_to_lds_b128 %0, %1, off"
                 :: "v"(lds_off), "v"((unsigned long long)(size_t)gaddr)
                 : "memory");
}
__device__ __forceinline__ void wait_async() {
    asm volatile("s_wait_asynccnt 0x0" ::: "memory");
}
__device__ __forceinline__ unsigned lds_off(const void* p) {
    return (unsigned)(size_t)p;  // generic LDS addr: low 32 bits = LDS offset
}

// ---------------------------------------------------------------------------
// fp32 -> bf16 conversion
// ---------------------------------------------------------------------------
__global__ void cvt_bf16_kernel(const float* __restrict__ in,
                                unsigned short* __restrict__ out, long n) {
    long i = (long)blockIdx.x * 256 + threadIdx.x;
    if (i < n) out[i] = f2bf(in[i]);
}

// Pack wq/wk/wv [H,D,DH] -> wqkv bf16 [D, 3*H*DH], and fused bias [3*H*DH]
__global__ void pack_qkv_kernel(const float* __restrict__ wq, const float* __restrict__ wk,
                                const float* __restrict__ wv, const float* __restrict__ bq,
                                const float* __restrict__ bk, const float* __restrict__ bv,
                                unsigned short* __restrict__ wqkv, float* __restrict__ bqkv) {
    long i = (long)blockIdx.x * 256 + threadIdx.x;
    const long N = 1024L * 3072L;
    if (i >= N) return;
    int d = (int)(i / 3072), c = (int)(i % 3072);
    int g = c / 1024, hh = (c % 1024) / 64, e = c % 64;
    const float* w = (g == 0) ? wq : (g == 1) ? wk : wv;
    wqkv[i] = f2bf(w[((long)hh * 1024 + d) * 64 + e]);
    if (d == 0) {
        const float* bb = (g == 0) ? bq : (g == 1) ? bk : bv;
        bqkv[c] = bb[hh * 64 + e];
    }
}

// ---------------------------------------------------------------------------
// Generic bf16 GEMM: C[M,N] = A[M,K]*B[K,N] (+bias)(+resid)(relu?)
// Block tile 256x64, K-step 32, 8 waves, wave tile 32x64 (8 WMMA / K-step).
// Double-buffered LDS; A tile via async global->LDS DMA, B via reg+transpose.
// ---------------------------------------------------------------------------
#define GBM 256
#define GBN 64
#define GBK 32
#define LDS32 40  // padded stride (u16) for 32-wide tiles: 80B, bank-spread, 16B-aligned

__global__ __launch_bounds__(256) void gemm_bf16_kernel(
    const unsigned short* __restrict__ A, int lda,
    const unsigned short* __restrict__ B, int ldb,
    const float* __restrict__ bias, const float* __restrict__ resid,
    unsigned short* __restrict__ outb, float* __restrict__ outf,
    int K, int ldc, int relu) {
    __shared__ __align__(16) unsigned short As[2][GBM * LDS32];  // 40 KB
    __shared__ __align__(16) unsigned short Bs[2][GBN * LDS32];  // 10 KB

    const int tid = threadIdx.x;
    const int wid = tid >> 5, lane = tid & 31;
    const int mrow = lane & 15, hh = lane >> 4;
    const long bm = (long)blockIdx.x * GBM;
    const long bn = (long)blockIdx.y * GBN;

    // A staging: 4x 16B chunks per thread, async DMA into LDS
    const unsigned short* aptr[4];
    unsigned aoff[4];
#pragma unroll
    for (int i = 0; i < 4; i++) {
        int v8 = tid + i * 256;        // 0..1023
        int m  = v8 >> 2;              // row 0..255
        int kk = (v8 & 3) << 3;        // 0,8,16,24
        aptr[i] = A + (bm + m) * (long)lda + kk;
        aoff[i] = lds_off(&As[0][m * LDS32 + kk]);
    }
    const unsigned abufstride = lds_off(&As[1][0]) - lds_off(&As[0][0]);

    // B staging: 1x 16B chunk per thread, transposed store to LDS
    const int bkk = tid >> 3;          // k row 0..31
    const int bnc = (tid & 7) << 3;    // n col 0,8,..,56
    const unsigned short* bptr = B + (long)bkk * ldb + bn + bnc;
    const long bstep = (long)GBK * ldb;

    f32x8 acc[2][4] = {};

    // prologue: tile 0 into buffer 0
#pragma unroll
    for (int i = 0; i < 4; i++) async_load_b128(aoff[i], aptr[i]);
    {
        u32x4 brq = *(const u32x4*)bptr;
        const unsigned short* bs = (const unsigned short*)&brq;
#pragma unroll
        for (int i = 0; i < 8; i++) Bs[0][(bnc + i) * LDS32 + bkk] = bs[i];
    }
    wait_async();
    __syncthreads();

    const int nk = K / GBK;
    int p = 0;
    for (int kt = 0; kt < nk; kt++) {
        const bool more = (kt + 1) < nk;
        u32x4 brn = {};
        if (more) {  // issue next tile: async A -> LDS, B -> regs
#pragma unroll
            for (int i = 0; i < 4; i++) {
                aptr[i] += GBK;
                async_load_b128(aoff[i] + (unsigned)(p ^ 1) * abufstride, aptr[i]);
            }
            bptr += bstep;
            brn = *(const u32x4*)bptr;
        }
        // compute current tile: wave rows [32*wid, 32*wid+32)
        Frag a0 = ldA(&As[p][(wid * 32 + mrow) * LDS32], hh);
        Frag a1 = ldA(&As[p][(wid * 32 + 16 + mrow) * LDS32], hh);
#pragma unroll
        for (int nt = 0; nt < 4; nt++) {
            Frag b = ldB(&Bs[p][(nt * 16 + mrow) * LDS32], hh);
            acc[0][nt] = WMMA_BF16(a0.v, b.v, acc[0][nt]);
            acc[1][nt] = WMMA_BF16(a1.v, b.v, acc[1][nt]);
        }
        if (more) {  // drain staged B into the other buffer
            const unsigned short* bs = (const unsigned short*)&brn;
#pragma unroll
            for (int i = 0; i < 8; i++) Bs[p ^ 1][(bnc + i) * LDS32 + bkk] = bs[i];
        }
        wait_async();      // this wave's async DMAs landed
        __syncthreads();   // all waves' stores visible
        p ^= 1;
    }

    // Epilogue: C fragment (r,lane) -> M = 32*wid + 16*ms + r + 8*hh, N = 16*nt + (lane&15)
#pragma unroll
    for (int ms = 0; ms < 2; ms++) {
#pragma unroll
        for (int nt = 0; nt < 4; nt++) {
#pragma unroll
            for (int r = 0; r < 8; r++) {
                long gm = bm + wid * 32 + ms * 16 + r + 8 * hh;
                long gn = bn + nt * 16 + mrow;
                float v = acc[ms][nt][r];
                if (bias) v += bias[gn];
                if (relu) v = fmaxf(v, 0.0f);
                if (outf) {
                    float rv = resid ? resid[gm * (long)ldc + gn] : 0.0f;
                    outf[gm * (long)ldc + gn] = v + rv;
                } else {
                    outb[gm * (long)ldc + gn] = f2bf(v);
                }
            }
        }
    }
}

// ---------------------------------------------------------------------------
// Causal flash attention. Block = (qtile of 128 rows, head, batch), 8 waves.
// Wave w owns 16 query rows -> softmax stats are wave-local.
// qkv layout: [B*S, 3*H*DH] bf16 (q | k | v). concat: [B*S, H*DH] bf16.
// ---------------------------------------------------------------------------
#define LDS64 72  // padded stride (u16) for 64-wide tiles: 144B, 16B-aligned

__global__ __launch_bounds__(256) void flash_attn_kernel(
    const unsigned short* __restrict__ qkv, unsigned short* __restrict__ concat) {
    const int S = 2048, H = 16, DH = 64, LDQ = 3 * H * DH;  // 3072
    __shared__ __align__(16) unsigned short Ks[64 * LDS64];     // [key][dh]
    __shared__ __align__(16) unsigned short Vt[64 * LDS64];     // [dh][key]
    __shared__ __align__(16) unsigned short Ps[8][16 * LDS64];  // per-wave P strip

    const int tid = threadIdx.x, wid = tid >> 5, lane = tid & 31;
    const int mrow = lane & 15, hh = lane >> 4;
    const int qt = blockIdx.x, h = blockIdx.y, b = blockIdx.z;
    const long rowbase = (long)b * S;
    const int q0 = qt * 128;
    const int colq = h * DH, colk = H * DH + h * DH, colv = 2 * H * DH + h * DH;

    // Q fragments (A layout) straight from global, kept in registers
    Frag aq0, aq1;
    {
        long r = rowbase + q0 + wid * 16 + mrow;
        const unsigned short* p = &qkv[r * LDQ + colq];
        aq0 = ldA(p, hh);
        aq1 = ldA(p + 32, hh);
    }

    // K-tile async staging (2x 16B per thread) and V-tile reg staging
    const int tkey = tid >> 3;           // 0..31 (keys; +32 for second chunk)
    const int tdh  = (tid & 7) << 3;     // 0,8,..,56
    const unsigned short* kptr[2];
    const unsigned short* vptr[2];
    unsigned koff[2];
#pragma unroll
    for (int i = 0; i < 2; i++) {
        int key = tkey + i * 32;
        kptr[i] = &qkv[(rowbase + key) * LDQ + colk + tdh];
        vptr[i] = &qkv[(rowbase + key) * LDQ + colv + tdh];
        koff[i] = lds_off(&Ks[key * LDS64 + tdh]);
    }
    const long jstep = (long)64 * LDQ;

    f32x8 acc[4] = {};
    float Mrun[8], Lrun[8];
#pragma unroll
    for (int r = 0; r < 8; r++) { Mrun[r] = -3.0e38f; Lrun[r] = 0.0f; }

    const int nj = 2 * qt + 2;  // key tiles of 64 covering the causal range
    for (int j = 0; j < nj; j++) {
        __syncthreads();  // prior iteration done reading Ks/Vt
#pragma unroll
        for (int i = 0; i < 2; i++) {
            async_load_b128(koff[i], kptr[i]);  // K tile via async DMA
            kptr[i] += jstep;
        }
#pragma unroll
        for (int i = 0; i < 2; i++) {           // V tile: vector load + transpose
            int key = tkey + i * 32;
            u32x4 vq = *(const u32x4*)vptr[i];
            vptr[i] += jstep;
            const unsigned short* vs = (const unsigned short*)&vq;
#pragma unroll
            for (int e = 0; e < 8; e++) Vt[(tdh + e) * LDS64 + key] = vs[e];
        }
        wait_async();
        __syncthreads();

        // scores = Q * K^T  (K-dim = DH = 64 -> two chained WMMAs)
        f32x8 sf[4];
#pragma unroll
        for (int nt = 0; nt < 4; nt++) {
            const unsigned short* kp = &Ks[(nt * 16 + mrow) * LDS64];
            Frag b0 = ldB(kp, hh), b1 = ldB(kp + 32, hh);
            f32x8 s = {};
            s = WMMA_BF16(aq0.v, b0.v, s);
            s = WMMA_BF16(aq1.v, b1.v, s);
            sf[nt] = s;
        }
        // scale 1/sqrt(DH) + causal mask
#pragma unroll
        for (int nt = 0; nt < 4; nt++) {
#pragma unroll
            for (int r = 0; r < 8; r++) {
                int qrow = q0 + wid * 16 + r + 8 * hh;
                int kcol = j * 64 + nt * 16 + mrow;
                float x = sf[nt][r] * 0.125f;
                sf[nt][r] = (kcol > qrow) ? -3.0e38f : x;
            }
        }
        // online softmax (wave-local): row max via 16-lane shuffle butterflies
        float alpha[8];
#pragma unroll
        for (int r = 0; r < 8; r++) {
            float v = fmaxf(fmaxf(sf[0][r], sf[1][r]), fmaxf(sf[2][r], sf[3][r]));
            v = fmaxf(v, __shfl_xor(v, 1, 32));
            v = fmaxf(v, __shfl_xor(v, 2, 32));
            v = fmaxf(v, __shfl_xor(v, 4, 32));
            v = fmaxf(v, __shfl_xor(v, 8, 32));
            float mnew = fmaxf(Mrun[r], v);
            alpha[r] = __expf(Mrun[r] - mnew);
            Mrun[r] = mnew;
        }
        // P = exp(s - m); row sums; rescale accumulators
#pragma unroll
        for (int r = 0; r < 8; r++) {
            float rs = 0.0f;
#pragma unroll
            for (int nt = 0; nt < 4; nt++) {
                float pe = __expf(sf[nt][r] - Mrun[r]);
                sf[nt][r] = pe;
                rs += pe;
            }
            rs += __shfl_xor(rs, 1, 32);
            rs += __shfl_xor(rs, 2, 32);
            rs += __shfl_xor(rs, 4, 32);
            rs += __shfl_xor(rs, 8, 32);
            Lrun[r] = Lrun[r] * alpha[r] + rs;
#pragma unroll
            for (int nt = 0; nt < 4; nt++) acc[nt][r] *= alpha[r];
        }
        // Re-swizzle P: C layout -> A layout via wave-private LDS strip
        unsigned short* pw = &Ps[wid][0];
#pragma unroll
        for (int nt = 0; nt < 4; nt++)
#pragma unroll
            for (int r = 0; r < 8; r++)
                pw[(r + 8 * hh) * LDS64 + nt * 16 + mrow] = f2bf(sf[nt][r]);
        asm volatile("s_wait_dscnt 0x0" ::: "memory");  // intra-wave LDS RAW
        Frag ap0 = ldA(&pw[mrow * LDS64], hh);
        Frag ap1 = ldA(&pw[mrow * LDS64] + 32, hh);
        // O += P * V  (K-dim = 64 keys -> two chained WMMAs per dh tile)
#pragma unroll
        for (int nt = 0; nt < 4; nt++) {
            const unsigned short* vp = &Vt[(nt * 16 + mrow) * LDS64];
            Frag b0 = ldB(vp, hh), b1 = ldB(vp + 32, hh);
            acc[nt] = WMMA_BF16(ap0.v, b0.v, acc[nt]);
            acc[nt] = WMMA_BF16(ap1.v, b1.v, acc[nt]);
        }
    }

    // normalize and store bf16 into concat [B*S, H*DH]
#pragma unroll
    for (int nt = 0; nt < 4; nt++) {
#pragma unroll
        for (int r = 0; r < 8; r++) {
            long row = rowbase + q0 + wid * 16 + r + 8 * hh;
            int cc = h * 64 + nt * 16 + mrow;
            concat[row * 1024 + cc] = f2bf(acc[nt][r] / Lrun[r]);
        }
    }
}

// ---------------------------------------------------------------------------
// LayerNorm: one block per row, D=1024. Writes f32 and/or bf16 outputs.
// ---------------------------------------------------------------------------
__global__ __launch_bounds__(256) void layernorm_kernel(
    const float* __restrict__ in, const float* __restrict__ sc,
    const float* __restrict__ bi, float* __restrict__ outf,
    unsigned short* __restrict__ outb, int D) {
    __shared__ float red[8];
    const long row = blockIdx.x;
    const float* x = in + row * (long)D;
    const int tid = threadIdx.x;

    float s = 0.0f;
    for (int i = tid; i < D; i += 256) s += x[i];
    for (int m = 16; m >= 1; m >>= 1) s += __shfl_xor(s, m, 32);
    if ((tid & 31) == 0) red[tid >> 5] = s;
    __syncthreads();
    float tot = 0.0f;
    for (int i = 0; i < 8; i++) tot += red[i];
    const float mean = tot / (float)D;
    __syncthreads();

    float vs = 0.0f;
    for (int i = tid; i < D; i += 256) { float d = x[i] - mean; vs += d * d; }
    for (int m = 16; m >= 1; m >>= 1) vs += __shfl_xor(vs, m, 32);
    if ((tid & 31) == 0) red[tid >> 5] = vs;
    __syncthreads();
    float v2 = 0.0f;
    for (int i = 0; i < 8; i++) v2 += red[i];
    const float rstd = rsqrtf(v2 / (float)D + 1e-5f);

    for (int i = tid; i < D; i += 256) {
        float y = (x[i] - mean) * rstd * sc[i] + bi[i];
        if (outf) outf[row * (long)D + i] = y;
        if (outb) outb[row * (long)D + i] = f2bf(y);
    }
}

// ---------------------------------------------------------------------------
// Host orchestration
// ---------------------------------------------------------------------------
extern "C" void kernel_launch(void* const* d_in, const int* in_sizes, int n_in,
                              void* d_out, int out_size, void* d_ws, size_t ws_size,
                              hipStream_t stream) {
    const int B = 4, S = 2048, D = 1024, H = 16, DFF = 4096;
    const long BS = (long)B * S;  // 8192

    const float* x    = (const float*)d_in[0];
    const float* wq   = (const float*)d_in[1];
    const float* bq   = (const float*)d_in[2];
    const float* wk   = (const float*)d_in[3];
    const float* bk   = (const float*)d_in[4];
    const float* wv   = (const float*)d_in[5];
    const float* bv   = (const float*)d_in[6];
    const float* wo   = (const float*)d_in[7];
    const float* bo   = (const float*)d_in[8];
    const float* ln1s = (const float*)d_in[9];
    const float* ln1b = (const float*)d_in[10];
    const float* w1   = (const float*)d_in[11];
    const float* b1   = (const float*)d_in[12];
    const float* w2   = (const float*)d_in[13];
    const float* b2   = (const float*)d_in[14];
    const float* ln2s = (const float*)d_in[15];
    const float* ln2b = (const float*)d_in[16];

    char* ws = (char*)d_ws;
    size_t off = 0;
    auto alloc = [&](size_t bytes) -> void* {
        void* p = ws + off;
        off += (bytes + 255) & ~(size_t)255;
        return p;
    };
    unsigned short* xb     = (unsigned short*)alloc(BS * D * 2);
    unsigned short* wqkvb  = (unsigned short*)alloc((size_t)D * 3072 * 2);
    float*          bqkv   = (float*)alloc(3072 * 4);
    unsigned short* qkvb   = (unsigned short*)alloc(BS * 3072 * 2);
    unsigned short* wob    = (unsigned short*)alloc((size_t)D * D * 2);
    unsigned short* w1b    = (unsigned short*)alloc((size_t)D * DFF * 2);
    unsigned short* w2b    = (unsigned short*)alloc((size_t)DFF * D * 2);
    unsigned short* concatb= (unsigned short*)alloc(BS * D * 2);
    float*          resid1 = (float*)alloc(BS * D * 4);
    float*          yf     = (float*)alloc(BS * D * 4);
    unsigned short* yb     = (unsigned short*)alloc(BS * D * 2);
    unsigned short* ff1b   = (unsigned short*)alloc(BS * DFF * 2);
    float*          resid2 = (float*)alloc(BS * D * 4);
    (void)ws_size; (void)in_sizes; (void)n_in; (void)out_size;

    // 1) conversions / packing
    {
        long n = BS * D;
        cvt_bf16_kernel<<<(unsigned)((n + 255) / 256), 256, 0, stream>>>(x, xb, n);
        long nw = (long)D * D;
        cvt_bf16_kernel<<<(unsigned)((nw + 255) / 256), 256, 0, stream>>>(wo, wob, nw);
        long n1 = (long)D * DFF;
        cvt_bf16_kernel<<<(unsigned)((n1 + 255) / 256), 256, 0, stream>>>(w1, w1b, n1);
        cvt_bf16_kernel<<<(unsigned)((n1 + 255) / 256), 256, 0, stream>>>(w2, w2b, n1);
        long np = (long)D * 3072;
        pack_qkv_kernel<<<(unsigned)((np + 255) / 256), 256, 0, stream>>>(
            wq, wk, wv, bq, bk, bv, wqkvb, bqkv);
    }
    // 2) fused QKV projection: [8192,1024] x [1024,3072] -> qkv bf16
    gemm_bf16_kernel<<<dim3(BS / GBM, 3072 / GBN), 256, 0, stream>>>(
        xb, D, wqkvb, 3072, bqkv, nullptr, qkvb, nullptr, D, 3072, 0);
    // 3) causal flash attention -> concat bf16 [8192,1024]
    flash_attn_kernel<<<dim3(S / 128, H, B), 256, 0, stream>>>(qkvb, concatb);
    // 4) output projection + bias + x residual -> resid1 f32
    gemm_bf16_kernel<<<dim3(BS / GBM, D / GBN), 256, 0, stream>>>(
        concatb, D, wob, D, bo, x, nullptr, resid1, D, D, 0);
    // 5) LayerNorm1 -> y (f32 + bf16)
    layernorm_kernel<<<(unsigned)BS, 256, 0, stream>>>(resid1, ln1s, ln1b, yf, yb, D);
    // 6) FFN up: relu(y*w1 + b1) -> ff1 bf16 [8192,4096]
    gemm_bf16_kernel<<<dim3(BS / GBM, DFF / GBN), 256, 0, stream>>>(
        yb, D, w1b, DFF, b1, nullptr, ff1b, nullptr, D, DFF, 1);
    // 7) FFN down + bias + y residual -> resid2 f32
    gemm_bf16_kernel<<<dim3(BS / GBM, D / GBN), 256, 0, stream>>>(
        ff1b, DFF, w2b, D, b2, yf, nullptr, resid2, DFF, D, 0);
    // 8) LayerNorm2 -> final f32 output
    layernorm_kernel<<<(unsigned)BS, 256, 0, stream>>>(
        resid2, ln2s, ln2b, (float*)d_out, nullptr, D);
}
